// NSMMPP_25512105739057
// MI455X (gfx1250) — matrix operand order repaired and checked by reading, compiled, not run on hardware
//
#include <hip/hip_runtime.h>
#include <hip/hip_bf16.h>

// ---------------- problem constants ----------------
#define Hdim   512
#define LNUM   32
#define NSEQ   16385
#define TSTEPS 16384          // N-1 scan steps
#define MSIM   16384
#define NEQ    7
#define ROWS   (NEQ*Hdim)     // 3584 stacked U rows
#define NWG    16             // persistent workgroups (== max cluster size)
#define CPW    (Hdim/NWG)     // 32 hidden columns per WG
#define RPW    (NEQ*CPW)      // 224 U rows per WG
#define TPW    (RPW/16)       // 14 WMMA row-tiles per WG
#define KCH    (Hdim/32)      // 16 K-chunks of 32

typedef __attribute__((ext_vector_type(16))) __bf16 v16bf;
typedef __attribute__((ext_vector_type(8)))  float  v8f;

// K index held by half `hh` of the A fragment for this lane
// (CDNA5 ISA 7.12.2, 16-bit A 16x32 layout)
__device__ __forceinline__ int kOfHalfA(int hh, int lane) {
  int kbase = (lane < 16) ? 0 : 8;
  int v = hh >> 1;
  int base = (v < 4) ? (2 * v) : (16 + 2 * (v - 4));
  return kbase + base + (hh & 1);
}

__device__ __forceinline__ float softplusf(float x) {
  return (x > 20.f) ? x : log1pf(expf(x));
}
__device__ __forceinline__ float sigmoidf(float x) {
  return 1.f / (1.f + expf(-x));
}

// device-wide sync for the persistent scan kernel:
// hardware cluster barrier when launched as a cluster, atomic fallback otherwise
__device__ __forceinline__ void gsync(unsigned* counter, unsigned* seq, bool inclu) {
  __syncthreads();
  __threadfence();
  if (inclu) {
    __builtin_amdgcn_s_cluster_barrier();   // s_barrier_signal -3 ; s_barrier_wait -3
  } else if (threadIdx.x == 0) {
    atomicAdd(counter, 1u);
    unsigned target = (*seq) * NWG;
    while (((volatile unsigned*)counter)[0] < target)
      __builtin_amdgcn_s_sleep(1);
  }
  __threadfence();
  ++(*seq);
  __syncthreads();
}

// ---------------------------------------------------------------------------
// Kernel 1: WEmbD[r][l] = sum_j Wall[r][j]*EmbExt[j][l] + d[r]
//           r in [0,3584), l in [0,34): 33 label columns + personalised vector
// One wave per 16x16 output tile, bf16 WMMA over K=512.
// ---------------------------------------------------------------------------
__global__ __launch_bounds__(128) void wemb_gemm(
    const float* __restrict__ W, const float* __restrict__ Emb,
    const float* __restrict__ EmbP, const int* __restrict__ seqIdPtr,
    const float* __restrict__ dvec, float* __restrict__ WEmbD)
{
  int lane = threadIdx.x & 31;
  int task = blockIdx.x * 4 + (threadIdx.x >> 5);   // 672 tasks = 224 x 3
  int tm = task / 3, tn = task % 3;
  int seq_id = *seqIdPtr;
  int rowM = lane & 15;
  int kbB  = (lane < 16) ? 0 : 16;

  v8f acc = {};
  for (int c = 0; c < KCH; ++c) {
    v16bf a, b;
    const float* Wrow = W + (size_t)(tm * 16 + rowM) * Hdim + c * 32;
    for (int hh = 0; hh < 16; ++hh)
      a[hh] = (__bf16)Wrow[kOfHalfA(hh, lane)];
    int col = tn * 16 + rowM;
    for (int hh = 0; hh < 16; ++hh) {
      int j = c * 32 + kOfHalfA(hh, lane);
      float v = 0.f;
      if (col < LNUM + 1)        v = Emb[(size_t)j * (LNUM + 1) + col];
      else if (col == LNUM + 1)  v = EmbP[(size_t)j * 1024 + seq_id];
      b[hh] = (__bf16)v;
    }
    acc = __builtin_amdgcn_wmma_f32_16x16x32_bf16(false, a, false, b,
                                                  (short)0, acc, false, false);
  }
  int col = tn * 16 + rowM;
  if (col < 34) {
    int rbase = tm * 16 + ((lane < 16) ? 0 : 8);
    for (int r = 0; r < 8; ++r) {
      int row = rbase + r;
      WEmbD[(size_t)row * 34 + col] = acc[r] + dvec[row];
    }
  }
}

// ---------------------------------------------------------------------------
// Kernel 2: persistent recurrent scan. 16 WGs x 224 threads (7 waves = 7 eqns).
// Each WG holds its 224x512 bf16 slice of U in LDS, pre-swizzled into WMMA-A
// fragments; per step: h -> bf16 LDS, 2 interleaved WMMA accumulation chains
// per wave (B = h broadcast into all 16 columns), gates on threads 0..31,
// ping-pong h buffers -> ONE device-wide sync per step.
// ---------------------------------------------------------------------------
#define SM_UFRAG   0                      // 14*16*32*16 halves = 229376 B
#define SM_HBF     229376                 // 512 bf16 = 1024 B
#define SM_WEMB    230400                 // 224*34 f32 = 30464 B
#define SM_TEMP    260864                 // 224 f32  = 896 B
#define SM_TOTAL   261760

__global__ __launch_bounds__(224, 1) void scan_kernel(
    const float* __restrict__ U, const int* __restrict__ label_seq,
    const float* __restrict__ time_seq, const float* __restrict__ w,
    const float* __restrict__ WEmbD,
    float* __restrict__ all_c, float* __restrict__ all_cb,
    float* __restrict__ all_delta, float* __restrict__ all_o,
    float* __restrict__ hbufs,            // 2 x Hdim floats, ping-pong
    float* __restrict__ hdot_part,
    unsigned* __restrict__ counter)
{
  extern __shared__ char smem[];
  __bf16* Ufrag = (__bf16*)(smem + SM_UFRAG);
  __bf16* hbf   = (__bf16*)(smem + SM_HBF);
  float*  LWE   = (float*) (smem + SM_WEMB);
  float*  temp  = (float*) (smem + SM_TEMP);

  const int tid  = threadIdx.x;
  const int g    = blockIdx.x;
  const int lane = tid & 31;
  const int wv   = tid >> 5;                       // wave == equation e
  const bool inclu = (__builtin_amdgcn_cluster_id_x() != 0);
  unsigned seq = 1;

  // ---- init: stage U slice into pre-swizzled WMMA-A fragment layout ----
  for (int trip = tid; trip < TPW * KCH * 32; trip += 224) {
    int ln = trip & 31;
    int c  = (trip >> 5) & 15;
    int ti = trip >> 9;
    int lr = ti * 16 + (ln & 15);
    int e = lr >> 5, ii = lr & 31;
    size_t gr = (size_t)e * Hdim + (size_t)g * CPW + ii;   // global U row
    const float* Urow = U + gr * Hdim + c * 32;
    v16bf pk;
    for (int hh = 0; hh < 16; ++hh)
      pk[hh] = (__bf16)Urow[((ln < 16) ? 0 : 16) + hh];
    *(v16bf*)&Ufrag[(size_t)((ti * KCH + c) * 32 + ln) * 16] = pk;
  }
  // stage this WG's slice of WEmbD (rows x 34 label columns)
  for (int x = tid; x < RPW * 34; x += 224) {
    int lr = x / 34, col = x % 34;
    int e = lr >> 5, ii = lr & 31;
    size_t gr = (size_t)e * Hdim + (size_t)g * CPW + ii;
    LWE[x] = WEmbD[gr * 34 + col];
  }

  float c_state = 0.f, cb_state = 0.f, w0 = 0.f;
  const int gi = g * CPW + lane;
  if (tid < 32) {
    hbufs[gi] = 0.f;                 // h_0 = 0 in buffer 0
    w0 = w[gi];                      // w row 0 (TARGET-1 == 0)
  }
  gsync(counter, &seq, inclu);

  const int kbB = (lane < 16) ? 0 : 16;
  const __bf16* ab0 = &Ufrag[(size_t)(wv * 2    ) * KCH * 32 * 16];
  const __bf16* ab1 = &Ufrag[(size_t)(wv * 2 + 1) * KCH * 32 * 16];

  // ---------------- the sequential scan ----------------
  for (int t = 0; t < TSTEPS; ++t) {
    const float* hr = hbufs + (size_t)(t & 1) * Hdim;         // read buffer
    float*       hw = hbufs + (size_t)((t + 1) & 1) * Hdim;   // write buffer

    // pull current h into LDS as bf16 (B operand source)
    for (int j = tid; j < Hdim; j += 224)
      hbf[j] = (__bf16)((volatile const float*)hr)[j];
    __syncthreads();

    // U.h for this WG's 224 rows: two interleaved accumulation chains
    v8f acc0 = {}, acc1 = {};
    for (int c = 0; c < KCH; ++c) {
      v16bf b  = *(const v16bf*)&hbf[c * 32 + kbB];  // h broadcast to 16 cols
      v16bf a0 = *(const v16bf*)&ab0[(size_t)(c * 32 + lane) * 16];
      v16bf a1 = *(const v16bf*)&ab1[(size_t)(c * 32 + lane) * 16];
      acc0 = __builtin_amdgcn_wmma_f32_16x16x32_bf16(false, a0, false, b,
                                                     (short)0, acc0, false, false);
      acc1 = __builtin_amdgcn_wmma_f32_16x16x32_bf16(false, a1, false, b,
                                                     (short)0, acc1, false, false);
    }
    if ((lane & 15) == 0) {                          // column 0 of D
      int half8 = (lane >> 4) << 3;
      int base0 = (wv * 2    ) * 16 + half8;
      int base1 = (wv * 2 + 1) * 16 + half8;
      for (int r = 0; r < 8; ++r) {
        temp[base0 + r] = acc0[r];
        temp[base1 + r] = acc1[r];
      }
    }
    __syncthreads();

    // gates + decay on threads 0..31 (one hidden column each)
    if (tid < 32) {
      int ii  = lane;
      int lab = (t == 0) ? (LNUM + 1) : label_seq[t];  // pers for step 0
      float tv[NEQ];
      for (int e = 0; e < NEQ; ++e)
        tv[e] = temp[e * 32 + ii] + LWE[(e * 32 + ii) * 34 + lab];
      float i_g = sigmoidf(tv[0]);
      float f_g = sigmoidf(tv[1]);
      float z   = tanhf(tv[2]);
      float o   = sigmoidf(tv[3]);
      float i2  = sigmoidf(tv[4]);
      float f2  = sigmoidf(tv[5]);
      float dl  = softplusf(tv[6]);
      float c   = f_g * c_state + i_g * z;
      cb_state  = f2 * cb_state + i2 * z;
      float tp = time_seq[t], tc = time_seq[t + 1];
      float ct = cb_state + (c - cb_state) * expf(-dl * (tc - tp));
      float h  = o * tanhf(ct);
      c_state = ct;

      size_t idx = (size_t)t * Hdim + gi;
      all_c[idx] = c; all_cb[idx] = cb_state; all_delta[idx] = dl; all_o[idx] = o;
      hw[gi] = h;

      float p = h * w0;                                // partial h . w[0]
      for (int off = 16; off > 0; off >>= 1) p += __shfl_xor(p, off, 32);
      if (lane == 0) hdot_part[(size_t)t * NWG + g] = p;
    }
    // single barrier per step: also guarantees everyone is done reading hr,
    // so hr may be overwritten (as hw) at step t+1.
    gsync(counter, &seq, inclu);
  }
}

// ---------------------------------------------------------------------------
// Kernel 3: Monte-Carlo term2. One wave per sim point.
// ---------------------------------------------------------------------------
__global__ __launch_bounds__(256) void sim_kernel(
    const int* __restrict__ sim_idx, const float* __restrict__ sim_time,
    const float* __restrict__ time_seq, const float* __restrict__ w,
    const float* __restrict__ log_s,
    const float* __restrict__ all_c, const float* __restrict__ all_cb,
    const float* __restrict__ all_delta, const float* __restrict__ all_o,
    float* __restrict__ term2_acc)
{
  int lane = threadIdx.x & 31;
  int m = blockIdx.x * 8 + (threadIdx.x >> 5);
  int idx = sim_idx[m];
  float dt = sim_time[m] - time_seq[idx];
  size_t base = (size_t)idx * Hdim;
  float s0 = expf(log_s[0]);
  float sum = 0.f;
  for (int jj = 0; jj < 16; ++jj) {
    int j = jj * 32 + lane;
    float cb = all_cb[base + j], cc = all_c[base + j];
    float dl = all_delta[base + j], oo = all_o[base + j];
    float cs = cb + (cc - cb) * expf(-dl * dt);
    float hs = oo * tanhf(cs);
    sum += hs * w[j];
  }
  for (int off = 16; off > 0; off >>= 1) sum += __shfl_xor(sum, off, 32);
  if (lane == 0) {
    float lam = s0 * softplusf(sum / s0) + 1e-9f;
    atomicAdd(term2_acc, lam);
  }
}

// ---------------------------------------------------------------------------
// Kernel 4: term1 reduction + final scalar
// ---------------------------------------------------------------------------
__global__ __launch_bounds__(256) void finalize_kernel(
    const float* __restrict__ hdot_part, const int* __restrict__ label_seq,
    const float* __restrict__ time_seq, const float* __restrict__ log_s,
    const float* __restrict__ term2_acc, float* __restrict__ out)
{
  __shared__ float red[256];
  float s0 = expf(log_s[0]);
  float local = 0.f;
  for (int t = threadIdx.x; t < TSTEPS - 1; t += 256) {   // h_obs rows
    if (label_seq[t + 1] == 1) {                          // TARGET == 1
      float hd = 0.f;
      for (int gq = 0; gq < NWG; ++gq) hd += hdot_part[(size_t)t * NWG + gq];
      local += logf(s0 * softplusf(hd / s0) + 1e-9f);
    }
  }
  red[threadIdx.x] = local;
  __syncthreads();
  for (int s = 128; s > 0; s >>= 1) {
    if (threadIdx.x < s) red[threadIdx.x] += red[threadIdx.x + s];
    __syncthreads();
  }
  if (threadIdx.x == 0) {
    float term1 = red[0];
    float term2 = (term2_acc[0] / (float)MSIM) * (time_seq[NSEQ - 1] - time_seq[0]);
    out[0] = -(term1 - term2);
  }
}

__global__ void init_kernel(float* term2_acc, unsigned* counter) {
  if (threadIdx.x == 0) { *term2_acc = 0.f; *counter = 0u; }
}

// ---------------------------------------------------------------------------
extern "C" void kernel_launch(void* const* d_in, const int* in_sizes, int n_in,
                              void* d_out, int out_size, void* d_ws, size_t ws_size,
                              hipStream_t stream)
{
  const int*   label_seq = (const int*)  d_in[0];
  const float* time_seq  = (const float*)d_in[1];
  const float* sim_time  = (const float*)d_in[2];
  const int*   sim_idx   = (const int*)  d_in[3];
  const int*   seq_id    = (const int*)  d_in[4];
  const float* Emb       = (const float*)d_in[5];
  const float* EmbP      = (const float*)d_in[6];
  const float* W         = (const float*)d_in[7];
  const float* U         = (const float*)d_in[8];
  const float* dvec      = (const float*)d_in[9];
  const float* w         = (const float*)d_in[10];
  const float* log_s     = (const float*)d_in[11];
  float* out = (float*)d_out;
  (void)in_sizes; (void)n_in; (void)out_size; (void)ws_size;

  char* ws = (char*)d_ws;
  size_t off = 0;
  auto alloc = [&](size_t bytes) -> char* {
    char* p = ws + off;
    off = (off + bytes + 255) & ~(size_t)255;
    return p;
  };
  float*    WEmbD     = (float*)   alloc((size_t)ROWS * 34 * 4);
  float*    all_c     = (float*)   alloc((size_t)TSTEPS * Hdim * 4);
  float*    all_cb    = (float*)   alloc((size_t)TSTEPS * Hdim * 4);
  float*    all_delta = (float*)   alloc((size_t)TSTEPS * Hdim * 4);
  float*    all_o     = (float*)   alloc((size_t)TSTEPS * Hdim * 4);
  float*    hbufs     = (float*)   alloc((size_t)2 * Hdim * 4);
  float*    hdot_part = (float*)   alloc((size_t)TSTEPS * NWG * 4);
  float*    term2_acc = (float*)   alloc(4);
  unsigned* counter   = (unsigned*)alloc(4);

  hipLaunchKernelGGL(init_kernel, dim3(1), dim3(64), 0, stream, term2_acc, counter);
  hipLaunchKernelGGL(wemb_gemm, dim3(168), dim3(128), 0, stream,
                     W, Emb, EmbP, seq_id, dvec, WEmbD);
  hipLaunchKernelGGL(scan_kernel, dim3(NWG), dim3(224), SM_TOTAL, stream,
                     U, label_seq, time_seq, w, WEmbD,
                     all_c, all_cb, all_delta, all_o, hbufs, hdot_part, counter);
  hipLaunchKernelGGL(sim_kernel, dim3(MSIM / 8), dim3(256), 0, stream,
                     sim_idx, sim_time, time_seq, w, log_s,
                     all_c, all_cb, all_delta, all_o, term2_acc);
  hipLaunchKernelGGL(finalize_kernel, dim3(1), dim3(256), 0, stream,
                     hdot_part, label_seq, time_seq, log_s, term2_acc, out);
}
